// MultiHeadAttention_3788161155688
// MI455X (gfx1250) — matrix operand order repaired and checked by reading, not compiled, not run on hardware
//
#include <hip/hip_runtime.h>

// ---------------------------------------------------------------------------
// MI455X / gfx1250 multi-head attention: bf16 WMMA pipeline, f32 accumulate.
//   1) gemm128<QK>  : q = Q@Wq+bq  -> bf16 [B,H,S,D]
//   2) gemm128<QK>  : k = K@Wk+bk  -> bf16 [B,H,S,D]
//   3) gemm128<VT>  : v = V@Wv+bv  -> bf16 [B,H,D,S]   (transposed for P@V)
//   4) attn_kernel  : flash attention, double-buffered async K/V staging
//   5) gemm128<OUT> : out = attn@Wo+bo -> f32 d_out
// ---------------------------------------------------------------------------

typedef __attribute__((ext_vector_type(16))) __bf16          v16bf;
typedef __attribute__((ext_vector_type(16))) unsigned short  v16u;
typedef __attribute__((ext_vector_type(8)))  unsigned short  v8u;
typedef __attribute__((ext_vector_type(8)))  float           v8f;
typedef __attribute__((ext_vector_type(4)))  int             v4i;

#define ATTN_B   2
#define ATTN_S   2048
#define ATTN_E   1024
#define ATTN_H   16
#define ATTN_D   64
#define ATTN_M   4096    /* B*S rows */

#define AS1 __attribute__((address_space(1)))
#define AS3 __attribute__((address_space(3)))

#if __has_builtin(__builtin_amdgcn_global_load_async_to_lds_b128) && \
    __has_builtin(__builtin_amdgcn_s_wait_asynccnt)
#define USE_ASYNC_LDS 1
#endif

// Direct memory->LDS async copy (16B per lane), ASYNCcnt-tracked on gfx1250.
// Builtin prototype (from hipcc diagnostic): (int4 AS1* src, int4 AS3* dst,
// imm offset, imm cpol).
__device__ __forceinline__ void stage_b128(void* lds, const void* gsrc) {
#ifdef USE_ASYNC_LDS
  __builtin_amdgcn_global_load_async_to_lds_b128(
      (AS1 v4i*)gsrc, (AS3 v4i*)lds, 0, 0);
#else
  *(uint4*)lds = *(const uint4*)gsrc;
#endif
}
__device__ __forceinline__ void stage_wait() {
#ifdef USE_ASYNC_LDS
  __builtin_amdgcn_s_wait_asynccnt(0);
#endif
}

__device__ __forceinline__ unsigned short f32_to_bf16(float f) {
  union { float f; unsigned u; } c; c.f = f;
  unsigned u = c.u;
  u += 0x7FFFu + ((u >> 16) & 1u);     // round-to-nearest-even
  return (unsigned short)(u >> 16);
}

__device__ __forceinline__ v8f vzero8() {
  v8f z;
#pragma unroll
  for (int e = 0; e < 8; ++e) z[e] = 0.0f;
  return z;
}

// WMMA wrapper: D = A(16x32 bf16) * B(32x16 bf16) + C(16x16 f32)
__device__ __forceinline__ v8f wmma_bf16(v16bf a, v16bf b, v8f c) {
  return __builtin_amdgcn_wmma_f32_16x16x32_bf16(
      /*neg_a=*/false, a, /*neg_b=*/false, b,
      /*c_mod=*/(short)0, c, /*reuse_a=*/false, /*reuse_b=*/false);
}

// A-fragment (16x32, bf16) from LDS tile with row stride 64 elements.
// Layout: lanes 0-15 hold row M=lane, K=0..15; lanes 16-31 hold
// row M=lane-16, K=16..31.  Two 16B ds reads per lane.
__device__ __forceinline__ v16bf load_a_frag(const unsigned short* p,
                                             int row, int kbase, int lane) {
  int hb = (lane < 16) ? 0 : 16;
  const unsigned short* r = p + row * 64 + kbase + hb;
  union { v8u h[2]; v16u w; } u;
  u.h[0] = *(const v8u*)(r);
  u.h[1] = *(const v8u*)(r + 8);
  return __builtin_bit_cast(v16bf, u.w);
}

// B-fragment (32x16, bf16): lane n = column n; lanes 0-15 hold K 0..15,
// lanes 16-31 hold K 16..31 -> one contiguous 32B read per lane.
__device__ __forceinline__ v16bf load_b_frag(const unsigned short* p) {
  v16u w = *(const v16u*)p;
  return __builtin_bit_cast(v16bf, w);
}

// ---------------------------------------------------------------------------
// 128x128-tile GEMM (8 waves, each wave = 16 rows x 128 cols strip):
//   Y[4096,1024] = X[4096,1024] @ W[1024,1024] + bias
// OMODE 0: store bf16 to [B,H,S,D]   (q / k projections)
// OMODE 2: store bf16 to [B,H,D,S]   (v projection, transposed per head)
// OMODE 3: store f32 row-major       (final output projection)
// ---------------------------------------------------------------------------
template <int OMODE, bool INBF16>
__global__ __launch_bounds__(256) void gemm128(const void* __restrict__ Xv,
                                               const float* __restrict__ W,
                                               const float* __restrict__ bias,
                                               void* __restrict__ outv) {
  constexpr int KD = 1024, N = 1024;
  __shared__ alignas(64) unsigned short xs[128 * 64];   // [m][k] bf16
  __shared__ alignas(64) unsigned short wt[128 * 64];   // [n][k] bf16 (W^T)

  const int t = threadIdx.x, lane = t & 31, wq = t >> 5;     // wq 0..7
  const int m0 = blockIdx.x * 128, n0 = blockIdx.y * 128;

  v8f acc[8];
#pragma unroll
  for (int i = 0; i < 8; ++i) acc[i] = vzero8();

  for (int kc = 0; kc < KD; kc += 64) {
    // ---- stage X tile: 128 rows x 64 k (convert to bf16) ----
    if (INBF16) {
      const unsigned short* X = (const unsigned short*)Xv;
#pragma unroll
      for (int i = 0; i < 4; ++i) {
        int vf = i * 256 + t;          // 1024 uint4 vectors total
        int r = vf >> 3, c = vf & 7;
        ((uint4*)xs)[vf] =
            *(const uint4*)(X + (size_t)(m0 + r) * KD + kc + c * 8);
      }
    } else {
      const float* X = (const float*)Xv;
#pragma unroll
      for (int i = 0; i < 32; ++i) {
        int f = i * 256 + t;           // 8192 elements
        int r = f >> 6, c = f & 63;
        xs[f] = f32_to_bf16(X[(size_t)(m0 + r) * KD + kc + c]);
      }
    }
    // ---- stage W tile transposed: wt[n][k], n in [0,128), k in [0,64) ----
#pragma unroll
    for (int i = 0; i < 32; ++i) {
      int f = i * 256 + t;             // 8192 elements
      int k = f >> 7, n = f & 127;     // coalesced along n in global
      wt[n * 64 + k] = f32_to_bf16(W[(size_t)(kc + k) * N + n0 + n]);
    }
    __syncthreads();

    const int row = wq * 16 + (lane & 15);
    const int hb2 = (lane < 16) ? 0 : 16;
#pragma unroll
    for (int kk = 0; kk < 64; kk += 32) {
      v16bf a = load_a_frag(xs, row, kk, lane);
#pragma unroll
      for (int nt = 0; nt < 8; ++nt) {
        v16bf b = load_b_frag(wt + (nt * 16 + (lane & 15)) * 64 + kk + hb2);
        acc[nt] = wmma_bf16(a, b, acc[nt]);
      }
    }
    __syncthreads();
  }

  // ---- epilogue: bias + layout-specific store ----
  const int halfo = (lane < 16) ? 0 : 8;
#pragma unroll
  for (int nt = 0; nt < 8; ++nt) {
    const int col = n0 + nt * 16 + (lane & 15);
    const float bv = bias[col];
#pragma unroll
    for (int j = 0; j < 8; ++j) {
      float val = acc[nt][j] + bv;
      int rowm = m0 + wq * 16 + j + halfo;
      if (OMODE == 3) {
        ((float*)outv)[(size_t)rowm * N + col] = val;
      } else {
        int bidx = rowm >> 11, s = rowm & 2047;   // S = 2048
        int h = col >> 6, d = col & 63;           // D = 64
        if (OMODE == 2)  // v^T : [B,H,D,S]
          ((unsigned short*)outv)[(((size_t)bidx * ATTN_H + h) * ATTN_D + d) *
                                      ATTN_S + s] = f32_to_bf16(val);
        else             // q/k : [B,H,S,D]
          ((unsigned short*)outv)[(((size_t)bidx * ATTN_H + h) * ATTN_S + s) *
                                      ATTN_D + d] = f32_to_bf16(val);
      }
    }
  }
}

// ---------------------------------------------------------------------------
// Flash attention: one block = 64 query rows of one (batch, head).
// 4 waves; each wave owns 16 query rows (Q frags resident in VGPRs).
// Double-buffered K/V tiles streamed via async global->LDS copies; per
// 64-key block: QK^T (8 wmma) -> online softmax -> P@V (8 wmma).
// ---------------------------------------------------------------------------
__global__ __launch_bounds__(128) void attn_kernel(
    const unsigned short* __restrict__ qb,
    const unsigned short* __restrict__ kbuf,
    const unsigned short* __restrict__ vtb,
    unsigned short* __restrict__ attn_out) {
  __shared__ alignas(64) unsigned short Qs[64 * 64];          // [row][d]
  __shared__ alignas(64) unsigned short Ks[2][64 * 64];       // [key][d]
  __shared__ alignas(64) unsigned short Vs[2][64 * 64];       // [d][key] (V^T)
  __shared__ alignas(64) unsigned short Ps[4][16 * 64];       // per-wave P

  const int t = threadIdx.x, lane = t & 31, wq = t >> 5;
  const int qblk = blockIdx.x;          // 0..31  (S/64)
  const int bh = blockIdx.y;            // 0..31  (B*H)
  const int hb2 = (lane < 16) ? 0 : 16;

  const unsigned short* kbase = kbuf + (size_t)bh * ATTN_S * ATTN_D;
  const unsigned short* vbase = vtb + (size_t)bh * ATTN_D * ATTN_S;

  // ---- prologue: stage Q tile + first K/V tiles into buffer 0 ----
  {
    const uint4* src = (const uint4*)(qb + (size_t)bh * ATTN_S * ATTN_D +
                                      (size_t)qblk * 64 * ATTN_D);
#pragma unroll
    for (int i = 0; i < 4; ++i)
      stage_b128(((uint4*)Qs) + t + 128 * i, src + t + 128 * i);
  }
  {
    const uint4* ksrc = (const uint4*)(kbase);         // kb = 0, contiguous
    const uint4* vsrc = (const uint4*)(vbase);         // kb = 0
#pragma unroll
    for (int i = 0; i < 4; ++i) {
      int f = t + 128 * i;
      stage_b128(((uint4*)Ks[0]) + f, ksrc + f);
      int r = f >> 3, c = f & 7;
      stage_b128(((uint4*)Vs[0]) + f, vsrc + r * (ATTN_S / 8) + c);
    }
  }
  stage_wait();
  __syncthreads();

  const int row = wq * 16 + (lane & 15);
  const v16bf qa0 = load_a_frag(Qs, row, 0, lane);
  const v16bf qa1 = load_a_frag(Qs, row, 32, lane);

  v8f o[4];
  float m[8], l[8];
#pragma unroll
  for (int i = 0; i < 4; ++i) o[i] = vzero8();
#pragma unroll
  for (int j = 0; j < 8; ++j) { m[j] = -1e30f; l[j] = 0.0f; }

  for (int kb = 0; kb < ATTN_S; kb += 64) {
    const int cur = (kb >> 6) & 1, nxt = cur ^ 1;
    // ---- prefetch next K/V tiles into the other buffer (no wait) ----
    if (kb + 64 < ATTN_S) {
      const uint4* ksrc = (const uint4*)(kbase + (size_t)(kb + 64) * ATTN_D);
      const uint4* vsrc = (const uint4*)(vbase + (size_t)(kb + 64));
#pragma unroll
      for (int i = 0; i < 4; ++i) {
        int f = t + 128 * i;
        stage_b128(((uint4*)Ks[nxt]) + f, ksrc + f);
        int r = f >> 3, c = f & 7;
        stage_b128(((uint4*)Vs[nxt]) + f, vsrc + r * (ATTN_S / 8) + c);
      }
    }

    // ---- S = (Q K^T) * 1/sqrt(D) ----
    v8f s[4];
#pragma unroll
    for (int i = 0; i < 4; ++i) s[i] = vzero8();
#pragma unroll
    for (int kk = 0; kk < 64; kk += 32) {
      v16bf a = (kk == 0) ? qa0 : qa1;
#pragma unroll
      for (int nt = 0; nt < 4; ++nt) {
        v16bf b =
            load_b_frag(Ks[cur] + (nt * 16 + (lane & 15)) * 64 + kk + hb2);
        s[nt] = wmma_bf16(a, b, s[nt]);
      }
    }
#pragma unroll
    for (int nt = 0; nt < 4; ++nt)
#pragma unroll
      for (int j = 0; j < 8; ++j) s[nt][j] *= 0.125f;   // 1/sqrt(64)

    // ---- online softmax: row max / rescale / exp / row sum ----
    float alpha[8];
#pragma unroll
    for (int j = 0; j < 8; ++j) {
      float mloc = fmaxf(fmaxf(s[0][j], s[1][j]), fmaxf(s[2][j], s[3][j]));
#pragma unroll
      for (int d = 1; d < 16; d <<= 1)
        mloc = fmaxf(mloc, __shfl_xor(mloc, d));
      float mn = fmaxf(m[j], mloc);
      alpha[j] = __expf(m[j] - mn);
      m[j] = mn;
    }
    float rs[8];
#pragma unroll
    for (int j = 0; j < 8; ++j) rs[j] = 0.0f;
    const int prow = (lane < 16) ? 0 : 8;
#pragma unroll
    for (int nt = 0; nt < 4; ++nt) {
      int colo = nt * 16 + (lane & 15);
#pragma unroll
      for (int j = 0; j < 8; ++j) {
        float p = __expf(s[nt][j] - m[j]);
        rs[j] += p;
        Ps[wq][(j + prow) * 64 + colo] = f32_to_bf16(p);
      }
    }
#pragma unroll
    for (int j = 0; j < 8; ++j) {
#pragma unroll
      for (int d = 1; d < 16; d <<= 1) rs[j] += __shfl_xor(rs[j], d);
      l[j] = l[j] * alpha[j] + rs[j];
    }
#pragma unroll
    for (int nt = 0; nt < 4; ++nt)
#pragma unroll
      for (int j = 0; j < 8; ++j) o[nt][j] *= alpha[j];

    __syncthreads();   // order P stores before P fragment reads

    // ---- O += P @ V ----
#pragma unroll
    for (int kk = 0; kk < 64; kk += 32) {
      v16bf pa = load_a_frag(Ps[wq], (lane & 15), kk, lane);
#pragma unroll
      for (int nt = 0; nt < 4; ++nt) {
        v16bf b =
            load_b_frag(Vs[cur] + (nt * 16 + (lane & 15)) * 64 + kk + hb2);
        o[nt] = wmma_bf16(pa, b, o[nt]);
      }
    }

    stage_wait();      // prefetched tiles landed in LDS
    __syncthreads();   // and everyone is done reading the current buffer
  }

  // ---- normalize and write bf16 [B,S,E] ----
  const int b_ = bh >> 4, h = bh & 15;
  const int halfo = (lane < 16) ? 0 : 8;
  float inv[8];
#pragma unroll
  for (int j = 0; j < 8; ++j) inv[j] = 1.0f / l[j];
#pragma unroll
  for (int nt = 0; nt < 4; ++nt) {
    int col = h * ATTN_D + nt * 16 + (lane & 15);
#pragma unroll
    for (int j = 0; j < 8; ++j) {
      int srow = qblk * 64 + wq * 16 + j + halfo;
      attn_out[((size_t)b_ * ATTN_S + srow) * ATTN_E + col] =
          f32_to_bf16(o[nt][j] * inv[j]);
    }
  }
}

// ---------------------------------------------------------------------------
extern "C" void kernel_launch(void* const* d_in, const int* in_sizes, int n_in,
                              void* d_out, int out_size, void* d_ws,
                              size_t ws_size, hipStream_t stream) {
  (void)in_sizes; (void)n_in; (void)out_size; (void)ws_size;
  const float* Q  = (const float*)d_in[0];
  const float* K  = (const float*)d_in[1];
  const float* V  = (const float*)d_in[2];
  const float* Wq = (const float*)d_in[3];
  const float* bq = (const float*)d_in[4];
  const float* Wk = (const float*)d_in[5];
  const float* bk = (const float*)d_in[6];
  const float* Wv = (const float*)d_in[7];
  const float* bv = (const float*)d_in[8];
  const float* Wo = (const float*)d_in[9];
  const float* bo = (const float*)d_in[10];

  const size_t SZ = (size_t)ATTN_M * ATTN_E;            // 4M elements
  unsigned short* qws  = (unsigned short*)d_ws;          //  8 MB
  unsigned short* kws  = qws  + SZ;                      //  8 MB
  unsigned short* vtws = kws  + SZ;                      //  8 MB
  unsigned short* aws  = vtws + SZ;                      //  8 MB

  dim3 ggrid(ATTN_M / 128, ATTN_E / 128);                // 32 x 8
  gemm128<0, false><<<ggrid, 256, 0, stream>>>(Q, Wq, bq, qws);
  gemm128<0, false><<<ggrid, 256, 0, stream>>>(K, Wk, bk, kws);
  gemm128<2, false><<<ggrid, 256, 0, stream>>>(V, Wv, bv, vtws);

  attn_kernel<<<dim3(ATTN_S / 64, ATTN_B * ATTN_H), 128, 0, stream>>>(
      qws, kws, vtws, aws);

  gemm128<3, true><<<ggrid, 256, 0, stream>>>(aws, Wo, bo, (float*)d_out);
}
